// MultiHeadAttention_35897336660576
// MI455X (gfx1250) — compile-verified, matrix operand order checked by reading, not run on hardware
//
#include <hip/hip_runtime.h>

// ---------------------------------------------------------------------------
// MI455X (gfx1250) multi-head causal attention with SwiGLU projections.
// Contractions on v_wmma_f32_16x16x32_f16 (wave32); all LDS tile feeds via
// the Tensor Data Mover (tensor_load_to_lds, TENSORcnt), double-buffered.
// ---------------------------------------------------------------------------

typedef __attribute__((ext_vector_type(16))) _Float16 v16h;
typedef __attribute__((ext_vector_type(8)))  _Float16 v8h;
typedef __attribute__((ext_vector_type(4)))  _Float16 v4h;
typedef __attribute__((ext_vector_type(8)))  float    v8f;
typedef __attribute__((ext_vector_type(4)))  unsigned int u32x4;
typedef __attribute__((ext_vector_type(8)))  int      i32x8;
typedef __attribute__((ext_vector_type(4)))  int      i32x4;

constexpr int kDModel = 1024;
constexpr int kHeads  = 16;
constexpr int kDk     = 64;
constexpr int kB      = 4;
constexpr int kS      = 2048;
constexpr int kM      = kB * kS;          // 8192 token rows

__device__ __forceinline__ v8f wmma16(v16h a, v16h b, v8f c) {
  return __builtin_amdgcn_wmma_f32_16x16x32_f16(false, a, false, b, (short)0, c,
                                                false, false);
}

// Load a 16x32 K-major fragment from LDS (A-layout; B uses identical
// addressing with lane = N-row).  ISA 7.12.2.
__device__ __forceinline__ v16h ldfrag(const _Float16* base, int ld, int lane) {
  const int r  = lane & 15;
  const int hf = (lane >> 4) << 3;              // 0 or 8
  const _Float16* p = base + r * ld + hf;
  union { v16h v; v8h h[2]; } u;
  u.h[0] = *(const v8h*)(p);
  u.h[1] = *(const v8h*)(p + 16);
  return u.v;
}

// ---------------------------------------------------------------------------
// TDM 2D tile load: global -> LDS.  Units of 4 bytes (data_size=2).
// D# packing per CDNA5 ISA ch.8 (groups 2/3 zero => 2D tile).
// This toolchain exposes the 6-arg builtin (extra i32x8 group, zero-filled).
// ---------------------------------------------------------------------------
__device__ __forceinline__ void tdm_load_2d(unsigned lds_off,
                                            const _Float16* gptr,
                                            unsigned tensor_d0,   // row len, u32 units
                                            unsigned tensor_d1,   // #rows in tensor
                                            unsigned stride0,     // row stride, u32 units
                                            unsigned tile_d0,     // tile width, u32 units
                                            unsigned tile_d1) {   // tile height (rows)
  const unsigned long long ga = (unsigned long long)(size_t)gptr;
  u32x4 g0;
  g0.x = 1u;                                        // count=1, user mode
  g0.y = lds_off;                                   // LDS byte address
  g0.z = (unsigned)ga;                              // global_addr[31:0]
  g0.w = ((unsigned)(ga >> 32) & 0x01FFFFFFu) | (2u << 30);  // addr[56:32] | type=2
  i32x8 g1;
  g1[0] = (int)(2u << 16);                          // data_size = 4B; mask/flags = 0
  g1[1] = (int)(tensor_d0 << 16);                   // tensor_dim0[15:0] << 16
  g1[2] = (int)((tensor_d0 >> 16) | (tensor_d1 << 16));
  g1[3] = (int)((tensor_d1 >> 16) | (tile_d0 << 16));
  g1[4] = (int)(tile_d1 & 0xFFFFu);                 // tile_dim1 | tile_dim2=0
  g1[5] = (int)stride0;                             // tensor_dim0_stride[31:0]
  g1[6] = 0;                                        // stride0 hi | stride1 lo
  g1[7] = 0;
  const i32x4 z4 = {0, 0, 0, 0};
  const i32x8 z8 = {0, 0, 0, 0, 0, 0, 0, 0};
  __builtin_amdgcn_tensor_load_to_lds(g0, g1, z4, z4, z8, 0);
}

__device__ __forceinline__ unsigned lds_off(const void* p) {
  return (unsigned)(size_t)p;                       // low 32 bits = LDS byte offset
}

// ---------------------------------------------------------------------------
// f32 -> f16 conversion (4 elements / thread)
// ---------------------------------------------------------------------------
__global__ void cvt_f32_f16(const float* __restrict__ src,
                            _Float16* __restrict__ dst, int n) {
  int i = (blockIdx.x * blockDim.x + threadIdx.x) * 4;
  if (i < n) {
    float4 f = *(const float4*)(src + i);
    v4h o;
    o.x = (_Float16)f.x; o.y = (_Float16)f.y;
    o.z = (_Float16)f.z; o.w = (_Float16)f.w;
    *(v4h*)(dst + i) = o;
  }
}

// ---------------------------------------------------------------------------
// SwiGLU projection GEMM:  out = silu(X@W1.T + b1) * (X@W2.T + b2)
// Tile 128x64, K-step 32, 8 waves (4M x 2N).  TDM double-buffered tiles.
// vmode 0: out = [b,h,s,d] (Q/K)   vmode 1: out = [b,h,d,s] (V transposed)
// ---------------------------------------------------------------------------
__global__ __launch_bounds__(256) void swiglu_gemm(
    const _Float16* __restrict__ X,
    const _Float16* __restrict__ W1,
    const _Float16* __restrict__ W2,
    const float* __restrict__ b1,
    const float* __restrict__ b2,
    _Float16* __restrict__ out, int vmode) {
  __shared__ _Float16 sA[2][128 * 32];   // 16 KB
  __shared__ _Float16 sB1[2][64 * 32];   //  8 KB
  __shared__ _Float16 sB2[2][64 * 32];   //  8 KB

  const int t = threadIdx.x;
  const int lane = t & 31, wid = t >> 5;
  const int waveM = wid >> 1, waveN = wid & 1;
  const int m0 = blockIdx.y * 128, n0 = blockIdx.x * 64;

  // Stage issue: A rows m0..m0+127, B rows n0..n0+63, k window = ik*32 halfs.
  // Row = 1024 halfs = 512 u32 units; 32-half window = 16 units.
  auto issue = [&](int ik, int buf) {
    const size_t kw = (size_t)ik * 32;
    tdm_load_2d(lds_off(&sA[buf][0]),  X  + (size_t)m0 * kDModel + kw,
                512, (unsigned)kM, 512, 16, 128);
    tdm_load_2d(lds_off(&sB1[buf][0]), W1 + (size_t)n0 * kDModel + kw,
                512, (unsigned)kDModel, 512, 16, 64);
    tdm_load_2d(lds_off(&sB2[buf][0]), W2 + (size_t)n0 * kDModel + kw,
                512, (unsigned)kDModel, 512, 16, 64);
  };

  v8f acc1[2][2] = {{{0}}}, acc2[2][2] = {{{0}}};

  if (wid == 0) issue(0, 0);
  for (int ik = 0; ik < kDModel / 32; ++ik) {
    const int buf = ik & 1;
    if (wid == 0) {
      if (ik + 1 < kDModel / 32) {
        issue(ik + 1, buf ^ 1);
        __builtin_amdgcn_s_wait_tensorcnt(3);   // current stage landed
      } else {
        __builtin_amdgcn_s_wait_tensorcnt(0);
      }
    }
    __syncthreads();

    v16h a[2], f1[2], f2[2];
    for (int mt = 0; mt < 2; ++mt)
      a[mt] = ldfrag(&sA[buf][(waveM * 32 + mt * 16) * 32], 32, lane);
    for (int nt = 0; nt < 2; ++nt) {
      f1[nt] = ldfrag(&sB1[buf][(waveN * 32 + nt * 16) * 32], 32, lane);
      f2[nt] = ldfrag(&sB2[buf][(waveN * 32 + nt * 16) * 32], 32, lane);
    }
    for (int mt = 0; mt < 2; ++mt)
      for (int nt = 0; nt < 2; ++nt) {
        acc1[mt][nt] = wmma16(a[mt], f1[nt], acc1[mt][nt]);
        acc2[mt][nt] = wmma16(a[mt], f2[nt], acc2[mt][nt]);
      }
    __syncthreads();   // buffer may be overwritten by next-next stage
  }

  // Epilogue: bias + silu * gate, scatter f16 to head layout.
  const int hi = lane >> 4, lanelo = lane & 15;
  for (int mt = 0; mt < 2; ++mt)
    for (int nt = 0; nt < 2; ++nt) {
      const int c = n0 + waveN * 32 + nt * 16 + lanelo;     // model col
      const float bb1 = b1[c], bb2 = b2[c];
      const int h = c >> 6, d = c & 63;
      for (int i = 0; i < 8; ++i) {
        const int m = m0 + waveM * 32 + mt * 16 + i + hi * 8;  // token row
        const float y1 = acc1[mt][nt][i] + bb1;
        const float y2 = acc2[mt][nt][i] + bb2;
        const float g  = (y1 / (1.0f + __expf(-y1))) * y2;
        const int b = m >> 11, s = m & (kS - 1);
        size_t idx = vmode == 0
            ? ((size_t)(b * kHeads + h) * kS + s) * kDk + d
            : ((size_t)(b * kHeads + h) * kDk + d) * kS + s;
        out[idx] = (_Float16)g;
      }
    }
}

// ---------------------------------------------------------------------------
// Flash attention, causal.  128 threads (4 waves) per 64 query rows of one
// (b,h); wave owns 16 rows.  K/V^T tiles stream via TDM, double-buffered;
// online softmax; P re-enters A-layout through per-wave LDS.
// ---------------------------------------------------------------------------
__global__ __launch_bounds__(128) void flash_attn(
    const _Float16* __restrict__ Qm,   // [BH, S, 64]
    const _Float16* __restrict__ Km,   // [BH, S, 64]
    const _Float16* __restrict__ VTm,  // [BH, 64, S]
    _Float16* __restrict__ O) {        // [B*S, 1024] f16
  __shared__ _Float16 sK[2][64 * 64];   // 16 KB
  __shared__ _Float16 sVT[2][64 * 64];  // 16 KB
  __shared__ _Float16 sP[4][16 * 64];   //  8 KB

  const int t = threadIdx.x, lane = t & 31, wid = t >> 5;
  const int hi = lane >> 4, lanelo = lane & 15;
  const int bh = blockIdx.y;
  const int b = bh >> 4, h = bh & 15;
  const int qt0 = blockIdx.x * 64;
  const int q0 = qt0 + wid * 16;

  const _Float16* Kb = Km  + (size_t)bh * kS * kDk;
  const _Float16* Vb = VTm + (size_t)bh * kDk * kS;

  // K tile: 64 rows x 64 halfs (row = 64 halfs = 32 u32 units, stride 32).
  // VT tile: 64 d-rows x 64 halfs, memory row stride = S halfs = 1024 units.
  auto issue = [&](int kb, int buf) {
    tdm_load_2d(lds_off(&sK[buf][0]),  Kb + (size_t)kb * 64 * kDk,
                32, (unsigned)kS, 32, 32, 64);
    tdm_load_2d(lds_off(&sVT[buf][0]), Vb + (size_t)kb * 64,
                1024, (unsigned)kDk, 1024, 32, 64);
  };

  // Q fragment (16 rows x 64 d) straight from global.
  v16h qf[2];
  {
    const _Float16* qrow = Qm + ((size_t)bh * kS + q0) * kDk;
    qf[0] = ldfrag(qrow, kDk, lane);
    qf[1] = ldfrag(qrow + 32, kDk, lane);
  }

  float mrow[8], lrow[8];
  v8f acco[4] = {{0}};
  for (int i = 0; i < 8; ++i) { mrow[i] = -3.0e38f; lrow[i] = 0.0f; }

  const int nkb = blockIdx.x + 1;          // causal: keys up to qt0+63
  if (wid == 0) issue(0, 0);
  for (int kb = 0; kb < nkb; ++kb) {
    const int kj = kb * 64;
    const int buf = kb & 1;
    if (wid == 0) {
      if (kb + 1 < nkb) {
        issue(kb + 1, buf ^ 1);
        __builtin_amdgcn_s_wait_tensorcnt(2);
      } else {
        __builtin_amdgcn_s_wait_tensorcnt(0);
      }
    }
    __syncthreads();

    // Scores: 4 tiles of 16q x 16k, contraction over d=64.
    v8f s4[4];
    for (int nt = 0; nt < 4; ++nt) {
      v8f z = {0};
      z = wmma16(qf[0], ldfrag(&sK[buf][(nt * 16) * 64], 64, lane), z);
      s4[nt] = wmma16(qf[1], ldfrag(&sK[buf][(nt * 16) * 64 + 32], 64, lane), z);
    }
    // Scale + causal mask (C layout: elem i -> q row i+8*hi, lanelo -> key).
    for (int nt = 0; nt < 4; ++nt)
      for (int i = 0; i < 8; ++i) {
        const int qg = q0 + i + hi * 8;
        const int kg = kj + nt * 16 + lanelo;
        float sv = s4[nt][i] * 0.125f;               // 1/sqrt(64)
        if (kg > qg) sv = -1.0e9f;
        s4[nt][i] = sv;
      }
    // Online softmax state update (reductions within 16-lane halves).
    float sc[8];
    for (int i = 0; i < 8; ++i) {
      float v = s4[0][i];
      v = fmaxf(v, s4[1][i]); v = fmaxf(v, s4[2][i]); v = fmaxf(v, s4[3][i]);
      for (int d = 1; d < 16; d <<= 1) v = fmaxf(v, __shfl_xor(v, d, 32));
      const float mn = fmaxf(mrow[i], v);
      sc[i] = __expf(mrow[i] - mn);
      mrow[i] = mn;
    }
    float rs[8] = {0, 0, 0, 0, 0, 0, 0, 0};
    for (int nt = 0; nt < 4; ++nt)
      for (int i = 0; i < 8; ++i) {
        const float p = __expf(s4[nt][i] - mrow[i]);
        rs[i] += p;
        sP[wid][(i + hi * 8) * 64 + nt * 16 + lanelo] = (_Float16)p;
      }
    for (int i = 0; i < 8; ++i) {
      float v = rs[i];
      for (int d = 1; d < 16; d <<= 1) v += __shfl_xor(v, d, 32);
      lrow[i] = lrow[i] * sc[i] + v;
      for (int nt = 0; nt < 4; ++nt) acco[nt][i] *= sc[i];
    }
    __syncthreads();   // order P stores before A-layout reload

    // PV: acc[16q x 64d] += P[16q x 64k] @ VT[64k x 64d]
    v16h pf0 = ldfrag(&sP[wid][0], 64, lane);
    v16h pf1 = ldfrag(&sP[wid][32], 64, lane);
    for (int nt = 0; nt < 4; ++nt) {
      acco[nt] = wmma16(pf0, ldfrag(&sVT[buf][(nt * 16) * 64], 64, lane), acco[nt]);
      acco[nt] = wmma16(pf1, ldfrag(&sVT[buf][(nt * 16) * 64 + 32], 64, lane), acco[nt]);
    }
    __syncthreads();   // buffer reuse fence
  }

  // Normalize and store merged-head f16 activations for the output GEMM.
  for (int nt = 0; nt < 4; ++nt)
    for (int i = 0; i < 8; ++i) {
      const int q = q0 + i + hi * 8;
      const int d = nt * 16 + lanelo;
      const float o = acco[nt][i] / lrow[i];
      O[((size_t)(b * kS + q)) * kDModel + h * kDk + d] = (_Float16)o;
    }
}

// ---------------------------------------------------------------------------
// Output projection: out_f32 = X_f16 @ Wo.T + bo   (TDM double-buffered)
// ---------------------------------------------------------------------------
__global__ __launch_bounds__(256) void out_proj(
    const _Float16* __restrict__ X,
    const _Float16* __restrict__ W,
    const float* __restrict__ bias,
    float* __restrict__ out) {
  __shared__ _Float16 sA[2][128 * 32];
  __shared__ _Float16 sB[2][64 * 32];

  const int t = threadIdx.x;
  const int lane = t & 31, wid = t >> 5;
  const int waveM = wid >> 1, waveN = wid & 1;
  const int m0 = blockIdx.y * 128, n0 = blockIdx.x * 64;

  auto issue = [&](int ik, int buf) {
    const size_t kw = (size_t)ik * 32;
    tdm_load_2d(lds_off(&sA[buf][0]), X + (size_t)m0 * kDModel + kw,
                512, (unsigned)kM, 512, 16, 128);
    tdm_load_2d(lds_off(&sB[buf][0]), W + (size_t)n0 * kDModel + kw,
                512, (unsigned)kDModel, 512, 16, 64);
  };

  v8f acc[2][2] = {{{0}}};
  if (wid == 0) issue(0, 0);
  for (int ik = 0; ik < kDModel / 32; ++ik) {
    const int buf = ik & 1;
    if (wid == 0) {
      if (ik + 1 < kDModel / 32) {
        issue(ik + 1, buf ^ 1);
        __builtin_amdgcn_s_wait_tensorcnt(2);
      } else {
        __builtin_amdgcn_s_wait_tensorcnt(0);
      }
    }
    __syncthreads();
    v16h a[2], f[2];
    for (int mt = 0; mt < 2; ++mt)
      a[mt] = ldfrag(&sA[buf][(waveM * 32 + mt * 16) * 32], 32, lane);
    for (int nt = 0; nt < 2; ++nt)
      f[nt] = ldfrag(&sB[buf][(waveN * 32 + nt * 16) * 32], 32, lane);
    for (int mt = 0; mt < 2; ++mt)
      for (int nt = 0; nt < 2; ++nt)
        acc[mt][nt] = wmma16(a[mt], f[nt], acc[mt][nt]);
    __syncthreads();
  }

  const int hi = lane >> 4, lanelo = lane & 15;
  for (int mt = 0; mt < 2; ++mt)
    for (int nt = 0; nt < 2; ++nt) {
      const int c = n0 + waveN * 32 + nt * 16 + lanelo;
      const float bb = bias[c];
      for (int i = 0; i < 8; ++i) {
        const int m = m0 + waveM * 32 + mt * 16 + i + hi * 8;
        out[(size_t)m * kDModel + c] = acc[mt][nt][i] + bb;
      }
    }
}

// ---------------------------------------------------------------------------
// Launch.  d_in: query,key,value,mask, wq1,bq1,wq2,bq2, wk1,bk1,wk2,bk2,
//                wv1,bv1,wv2,bv2, wo,bo.   mask unused (causal is exact).
// Workspace (~126 MB): xq,xk,xv | 7 weights f16 | q16,k16,vT16,attn16.
// ---------------------------------------------------------------------------
extern "C" void kernel_launch(void* const* d_in, const int* in_sizes, int n_in,
                              void* d_out, int out_size, void* d_ws,
                              size_t ws_size, hipStream_t stream) {
  (void)in_sizes; (void)n_in; (void)out_size; (void)ws_size;
  const size_t ACT = (size_t)kM * kDModel;       // 8,388,608 elems
  const size_t WSZ = (size_t)kDModel * kDModel;  // 1,048,576 elems

  _Float16* p   = (_Float16*)d_ws;
  _Float16* xq  = p;              p += ACT;
  _Float16* xk  = p;              p += ACT;
  _Float16* xv  = p;              p += ACT;
  _Float16* wt  = p;              p += 7 * WSZ;  // wq1,wq2,wk1,wk2,wv1,wv2,wo
  _Float16* q16 = p;              p += ACT;
  _Float16* k16 = p;              p += ACT;
  _Float16* vT  = p;              p += ACT;
  _Float16* att = p;              p += ACT;

  // 1) f32 -> f16 conversions
  cvt_f32_f16<<<(int)(ACT / 1024), 256, 0, stream>>>((const float*)d_in[0], xq, (int)ACT);
  cvt_f32_f16<<<(int)(ACT / 1024), 256, 0, stream>>>((const float*)d_in[1], xk, (int)ACT);
  cvt_f32_f16<<<(int)(ACT / 1024), 256, 0, stream>>>((const float*)d_in[2], xv, (int)ACT);
  const int widx[7] = {4, 6, 8, 10, 12, 14, 16};
  for (int i = 0; i < 7; ++i)
    cvt_f32_f16<<<(int)(WSZ / 1024), 256, 0, stream>>>(
        (const float*)d_in[widx[i]], wt + (size_t)i * WSZ, (int)WSZ);

  // 2) SwiGLU projections
  dim3 ggrid(kDModel / 64, kM / 128);
  swiglu_gemm<<<ggrid, 256, 0, stream>>>(xq, wt + 0 * WSZ, wt + 1 * WSZ,
      (const float*)d_in[5], (const float*)d_in[7], q16, 0);
  swiglu_gemm<<<ggrid, 256, 0, stream>>>(xk, wt + 2 * WSZ, wt + 3 * WSZ,
      (const float*)d_in[9], (const float*)d_in[11], k16, 0);
  swiglu_gemm<<<ggrid, 256, 0, stream>>>(xv, wt + 4 * WSZ, wt + 5 * WSZ,
      (const float*)d_in[13], (const float*)d_in[15], vT, 1);

  // 3) Causal flash attention
  flash_attn<<<dim3(kS / 64, kB * kHeads), 128, 0, stream>>>(q16, k16, vT, att);

  // 4) Output projection (f32 result)
  out_proj<<<ggrid, 256, 0, stream>>>(att, wt + 6 * WSZ, (const float*)d_in[17],
                                      (float*)d_out);
}
